// PointCloudVQVAE_28836410425604
// MI455X (gfx1250) — matrix operand order repaired and checked by reading, not compiled, not run on hardware
//
#include <hip/hip_runtime.h>
#include <hip/hip_bf16.h>
#include <math.h>

typedef __attribute__((ext_vector_type(16))) _Float16 v16h;
typedef __attribute__((ext_vector_type(8)))  float    v8f;

#define LN_EPS 1e-5f

// ---------- helpers ----------
__device__ __forceinline__ float gelu_exact(float x) {
  return 0.5f * x * (1.0f + erff(x * 0.7071067811865475f));
}

// order-preserving float <-> uint mapping (works for min and max atomics)
__device__ __forceinline__ unsigned f2ord(float f) {
  unsigned u = __float_as_uint(f);
  return (u & 0x80000000u) ? ~u : (u | 0x80000000u);
}
__device__ __forceinline__ float ord2f(unsigned u) {
  unsigned b = (u & 0x80000000u) ? (u & 0x7FFFFFFFu) : ~u;
  return __uint_as_float(b);
}

union frag16 { uint4 q[2]; v16h v; };
union pk2 { _Float16 h[2]; unsigned u; };

// ---------- wave-level 16x16 WMMA tile GEMM: out = A(MxK) * B(KxN) + bias ----------
// A : LDS f16 row-major [*, lda] (lda multiple of 8)
// Bt: LDS f16 TRANSPOSED [N, K]  (so a B fragment is two contiguous 16B runs)
// out: LDS f32 [*, ldo]
__device__ __forceinline__ void wave_gemm_tile(const _Float16* A, int lda,
                                               const _Float16* Bt, int K,
                                               float* out, int ldo,
                                               const float* bias,
                                               int m0, int n0, int lane) {
  const int hi  = lane >> 4;   // lane half (0/1)
  const int l15 = lane & 15;
  v8f c = {};
  const _Float16* arow = A  + (size_t)(m0 + l15) * lda + 8 * hi;
  const _Float16* brow = Bt + (size_t)(n0 + l15) * K   + 8 * hi;
#pragma unroll
  for (int k0 = 0; k0 < 128; k0 += 32) {   // upper bound; break on K
    if (k0 >= K) break;
    frag16 a, b;
    // A fragment: lane holds row m0+l15.
    //   VGPR0..3 = K = k0+8*hi+0..7  (16B contiguous)
    //   VGPR4..7 = K = k0+8*hi+16..23 (16B contiguous)
    a.q[0] = *(const uint4*)(arow + k0);
    a.q[1] = *(const uint4*)(arow + k0 + 16);
    // B fragment: lane holds col n0+l15; K = k0+8*hi+0..7, k0+8*hi+16..23 (same as A)
    b.q[0] = *(const uint4*)(brow + k0);
    b.q[1] = *(const uint4*)(brow + k0 + 16);
    c = __builtin_amdgcn_wmma_f32_16x16x32_f16(false, a.v, false, b.v, (short)0,
                                               c, false, false);
  }
  // C/D layout: lane holds col n0+l15, VGPR r -> row m0 + 8*hi + r
  const int col = n0 + l15;
  const float bv = bias[col];
#pragma unroll
  for (int r = 0; r < 8; ++r) {
    out[(size_t)(m0 + 8 * hi + r) * ldo + col] = c[r] + bv;
  }
}

// LayerNorm + exact GELU over rows of F features; one thread per row. f16 out.
__device__ __forceinline__ void ln_gelu_rows_f16(const float* in, int F,
                                                 _Float16* out, const float* g,
                                                 const float* be, int t, int rows) {
  if (t < rows) {
    const float4* r4 = (const float4*)(in + (size_t)t * F);
    const int F4 = F >> 2;
    float mu = 0.f;
    for (int j = 0; j < F4; ++j) {
      float4 v = r4[j];
      mu += v.x + v.y + v.z + v.w;
    }
    mu /= (float)F;
    float var = 0.f;
    for (int j = 0; j < F4; ++j) {
      float4 v = r4[j];
      float d0 = v.x - mu, d1 = v.y - mu, d2 = v.z - mu, d3 = v.w - mu;
      var += d0 * d0 + d1 * d1 + d2 * d2 + d3 * d3;
    }
    var /= (float)F;
    const float s = rsqrtf(var + LN_EPS);
    unsigned* o2 = (unsigned*)(out + (size_t)t * F);
    const float* r = in + (size_t)t * F;
    for (int j = 0; j < F; j += 2) {
      pk2 p;
      p.h[0] = (_Float16)gelu_exact((r[j]     - mu) * s * g[j]     + be[j]);
      p.h[1] = (_Float16)gelu_exact((r[j + 1] - mu) * s * g[j + 1] + be[j + 1]);
      o2[j >> 1] = p.u;
    }
  }
}

// LayerNorm + GELU in place (f32 out).
__device__ __forceinline__ void ln_gelu_rows_f32(float* buf, int F, const float* g,
                                                 const float* be, int t, int rows) {
  if (t < rows) {
    float4* r4 = (float4*)(buf + (size_t)t * F);
    const int F4 = F >> 2;
    float mu = 0.f;
    for (int j = 0; j < F4; ++j) {
      float4 v = r4[j];
      mu += v.x + v.y + v.z + v.w;
    }
    mu /= (float)F;
    float var = 0.f;
    for (int j = 0; j < F4; ++j) {
      float4 v = r4[j];
      float d0 = v.x - mu, d1 = v.y - mu, d2 = v.z - mu, d3 = v.w - mu;
      var += d0 * d0 + d1 * d1 + d2 * d2 + d3 * d3;
    }
    var /= (float)F;
    const float s = rsqrtf(var + LN_EPS);
    for (int j = 0; j < F4; ++j) {
      float4 v = r4[j];
      int j4 = j * 4;
      v.x = gelu_exact((v.x - mu) * s * g[j4]     + be[j4]);
      v.y = gelu_exact((v.y - mu) * s * g[j4 + 1] + be[j4 + 1]);
      v.z = gelu_exact((v.z - mu) * s * g[j4 + 2] + be[j4 + 2]);
      v.w = gelu_exact((v.w - mu) * s * g[j4 + 3] + be[j4 + 3]);
      r4[j] = v;
    }
  }
}

// ---------- kernel 0: init accumulators ----------
__global__ void vq_init_kernel(unsigned* pooled, unsigned long long* key) {
  int t = threadIdx.x;
  pooled[t] = 0x007FFFFFu;  // f2ord(-inf)
  if (t == 0) *key = ~0ULL;
}

// ---------- kernel 1: fused encoder (3->64->128->256) + max-pool ----------
#define MT 64
__global__ void __launch_bounds__(256)
vq_encoder_kernel(const float* __restrict__ pts, int N,
                  const float* eW0, const float* eb0, const float* eg0, const float* ebeta0,
                  const float* eW1, const float* eb1, const float* eg1, const float* ebeta1,
                  const float* eW2, const float* eb2, const float* eg2, const float* ebeta2,
                  unsigned* __restrict__ pooled) {
  extern __shared__ char smem[];
  _Float16* W1t = (_Float16*)(smem);                // [128][64]  f16 = 16 KB (transposed)
  _Float16* W2t = (_Float16*)(smem + 16 * 1024);    // [256][128] f16 = 64 KB (transposed)
  float*    hf  = (float*)(smem + 80 * 1024);       // up to 64x256 f32 = 64 KB (reused)
  _Float16* A0  = (_Float16*)(smem + 144 * 1024);   //  64x64  f16 =  8 KB
  _Float16* A1  = (_Float16*)(smem + 152 * 1024);   //  64x128 f16 = 16 KB

  const int t = threadIdx.x;
  const int base = blockIdx.x * MT;

  // stage weights transposed (global coalesced read, strided LDS write), f32->f16
  for (int i = t; i < 64 * 128; i += 256) {
    int k = i >> 7, n = i & 127;          // eW1 is [64][128]
    W1t[n * 64 + k] = (_Float16)eW1[i];
  }
  for (int i = t; i < 128 * 256; i += 256) {
    int k = i >> 8, n = i & 255;          // eW2 is [128][256]
    W2t[n * 128 + k] = (_Float16)eW2[i];
  }

  // ---- layer 0: 3 -> 64 (VALU; 4 threads per point, 16 outputs each) ----
  {
    const int p  = t >> 2;
    const int o0 = (t & 3) * 16;
    const int gp = base + p;
    float x0 = 0.f, x1 = 0.f, x2 = 0.f;
    if (gp < N) { x0 = pts[gp * 3 + 0]; x1 = pts[gp * 3 + 1]; x2 = pts[gp * 3 + 2]; }
#pragma unroll
    for (int j = 0; j < 16; ++j) {
      int o = o0 + j;
      hf[p * 64 + o] = eb0[o] + x0 * eW0[0 * 64 + o] + x1 * eW0[1 * 64 + o]
                              + x2 * eW0[2 * 64 + o];
    }
  }
  __syncthreads();
  ln_gelu_rows_f16(hf, 64, A0, eg0, ebeta0, t, MT);
  __syncthreads();

  // ---- layer 1: 64 -> 128 (WMMA), 32 tiles over 8 waves ----
  {
    const int wave = t >> 5, lane = t & 31;
#pragma unroll
    for (int i = 0; i < 4; ++i) {
      const int tid = wave * 4 + i;
      wave_gemm_tile(A0, 64, W1t, 64, hf, 128, eb1,
                     (tid >> 3) * 16, (tid & 7) * 16, lane);
    }
  }
  __syncthreads();
  ln_gelu_rows_f16(hf, 128, A1, eg1, ebeta1, t, MT);
  __syncthreads();

  // ---- layer 2: 128 -> 256 (WMMA), 64 tiles over 8 waves ----
  {
    const int wave = t >> 5, lane = t & 31;
#pragma unroll
    for (int i = 0; i < 8; ++i) {
      const int tid = wave * 8 + i;
      wave_gemm_tile(A1, 128, W2t, 128, hf, 256, eb2,
                     (tid >> 4) * 16, (tid & 15) * 16, lane);
    }
  }
  __syncthreads();
  ln_gelu_rows_f32(hf, 256, eg2, ebeta2, t, MT);  // in place
  __syncthreads();

  // ---- masked column max + global ordered atomic max ----
  {
    const int rows = min(MT, N - base);
    float m = -3.4e38f;
    for (int r = 0; r < rows; ++r) m = fmaxf(m, hf[(size_t)r * 256 + t]);
    atomicMax(&pooled[t], f2ord(m));
  }
}

// ---------- kernel 2: z_e = pooled @ oW + ob ----------
__global__ void __launch_bounds__(256)
vq_latent_kernel(const unsigned* __restrict__ pooled, const float* __restrict__ oW,
                 const float* __restrict__ ob, float* __restrict__ z_e_out) {
  __shared__ float p[256];
  const int t = threadIdx.x;
  p[t] = ord2f(pooled[t]);
  __syncthreads();
  float acc = ob[t];
  for (int i = 0; i < 256; ++i) acc += p[i] * oW[i * 256 + t];
  z_e_out[t] = acc;
}

// ---------- kernel 3: codebook argmin (||c||^2 - 2 c.z; ||z||^2 const) ----------
__global__ void __launch_bounds__(256)
vq_argmin_kernel(const float* __restrict__ z_e, const float* __restrict__ codebook,
                 unsigned long long* __restrict__ key) {
  __shared__ float z[256];
  __shared__ unsigned long long red[256];
  const int t = threadIdx.x;
  z[t] = z_e[t];
  __syncthreads();
  const int row = blockIdx.x * 256 + t;
  const float* c = codebook + (size_t)row * 256;
  float cc = 0.f, cz = 0.f;
  for (int i = 0; i < 256; ++i) { float v = c[i]; cc += v * v; cz += v * z[i]; }
  const float dist = cc - 2.f * cz;
  red[t] = ((unsigned long long)f2ord(dist) << 32) | (unsigned)row;
  __syncthreads();
  for (int s = 128; s > 0; s >>= 1) {
    if (t < s) red[t] = red[t] < red[t + s] ? red[t] : red[t + s];
    __syncthreads();
  }
  if (t == 0) atomicMin(key, red[0]);
}

// ---------- kernel 4: gather z_q + decoder MLP (256->512->512->512) ----------
__device__ __forceinline__ float block_sum512(float v, float* scratch, int t) {
  scratch[t] = v;
  __syncthreads();
  for (int s = 256; s > 0; s >>= 1) {
    if (t < s) scratch[t] += scratch[t + s];
    __syncthreads();
  }
  float r = scratch[0];
  __syncthreads();
  return r;
}

__global__ void __launch_bounds__(512)
vq_decoder_mlp_kernel(const unsigned long long* __restrict__ key,
                      const float* __restrict__ codebook, float* __restrict__ z_q_out,
                      const float* dW0, const float* db0, const float* dg0, const float* dbeta0,
                      const float* dW1, const float* db1, const float* dg1, const float* dbeta1,
                      const float* dW2, const float* db2, const float* dg2, const float* dbeta2,
                      float* __restrict__ g_out) {
  __shared__ float zq[256];
  __shared__ float act[512];
  __shared__ float scratch[512];
  const int t = threadIdx.x;
  const int idx = (int)(*key & 0xFFFFFFFFull);
  if (t < 256) {
    float v = codebook[(size_t)idx * 256 + t];
    zq[t] = v;
    z_q_out[t] = v;
  }
  __syncthreads();

  // layer d0: 256 -> 512
  float a = db0[t];
  for (int i = 0; i < 256; ++i) a += zq[i] * dW0[i * 512 + t];
  {
    float mu = block_sum512(a, scratch, t) * (1.f / 512.f);
    float d = a - mu;
    float var = block_sum512(d * d, scratch, t) * (1.f / 512.f);
    act[t] = gelu_exact(d * rsqrtf(var + LN_EPS) * dg0[t] + dbeta0[t]);
  }
  __syncthreads();

  // layer d1: 512 -> 512
  a = db1[t];
  for (int i = 0; i < 512; ++i) a += act[i] * dW1[i * 512 + t];
  __syncthreads();
  {
    float mu = block_sum512(a, scratch, t) * (1.f / 512.f);
    float d = a - mu;
    float var = block_sum512(d * d, scratch, t) * (1.f / 512.f);
    act[t] = gelu_exact(d * rsqrtf(var + LN_EPS) * dg1[t] + dbeta1[t]);
  }
  __syncthreads();

  // layer d2: 512 -> 512
  a = db2[t];
  for (int i = 0; i < 512; ++i) a += act[i] * dW2[i * 512 + t];
  __syncthreads();
  {
    float mu = block_sum512(a, scratch, t) * (1.f / 512.f);
    float d = a - mu;
    float var = block_sum512(d * d, scratch, t) * (1.f / 512.f);
    g_out[t] = gelu_exact(d * rsqrtf(var + LN_EPS) * dg2[t] + dbeta2[t]);
  }
}

// ---------- kernel 5: recon = g @ doW + dob  (memory bound, coalesced cols) ----------
__global__ void __launch_bounds__(256)
vq_final_kernel(const float* __restrict__ g, const float* __restrict__ doW,
                const float* __restrict__ dob, float* __restrict__ recon, int KO) {
  __shared__ float gs[512];
  const int t = threadIdx.x;
  gs[t] = g[t];
  gs[t + 256] = g[t + 256];
  __syncthreads();
  const int j = blockIdx.x * 256 + t;
  float acc = dob[j];
  for (int i = 0; i < 512; ++i) acc += gs[i] * doW[(size_t)i * KO + j];
  recon[j] = acc;
}

// ---------- launch ----------
extern "C" void kernel_launch(void* const* d_in, const int* in_sizes, int n_in,
                              void* d_out, int out_size, void* d_ws, size_t ws_size,
                              hipStream_t stream) {
  const float* points  = (const float*)d_in[0];
  const float* eW0 = (const float*)d_in[1];  const float* eb0 = (const float*)d_in[2];
  const float* eg0 = (const float*)d_in[3];  const float* ebt0 = (const float*)d_in[4];
  const float* eW1 = (const float*)d_in[5];  const float* eb1 = (const float*)d_in[6];
  const float* eg1 = (const float*)d_in[7];  const float* ebt1 = (const float*)d_in[8];
  const float* eW2 = (const float*)d_in[9];  const float* eb2 = (const float*)d_in[10];
  const float* eg2 = (const float*)d_in[11]; const float* ebt2 = (const float*)d_in[12];
  const float* oW  = (const float*)d_in[13]; const float* ob  = (const float*)d_in[14];
  const float* codebook = (const float*)d_in[15];
  const float* dW0 = (const float*)d_in[16]; const float* db0 = (const float*)d_in[17];
  const float* dg0 = (const float*)d_in[18]; const float* dbt0 = (const float*)d_in[19];
  const float* dW1 = (const float*)d_in[20]; const float* db1 = (const float*)d_in[21];
  const float* dg1 = (const float*)d_in[22]; const float* dbt1 = (const float*)d_in[23];
  const float* dW2 = (const float*)d_in[24]; const float* db2 = (const float*)d_in[25];
  const float* dg2 = (const float*)d_in[26]; const float* dbt2 = (const float*)d_in[27];
  const float* doW = (const float*)d_in[28]; const float* dob = (const float*)d_in[29];

  const int N  = in_sizes[0] / 3;       // 500000
  const int KO = in_sizes[29];          // 24576 = K*3

  float* out     = (float*)d_out;
  float* recon   = out;                 // [0, KO)
  float* z_e_out = out + KO;            // [KO, KO+256)
  float* z_q_out = out + KO + 256;      // [KO+256, KO+512)

  char* ws = (char*)d_ws;
  unsigned* pooled          = (unsigned*)ws;                       // 256 u32
  unsigned long long* key   = (unsigned long long*)(ws + 1024);    // 1 u64
  float* g_ws               = (float*)(ws + 2048);                 // 512 f32

  vq_init_kernel<<<1, 256, 0, stream>>>(pooled, key);

  const int nb = (N + MT - 1) / MT;
  const size_t lds_bytes = 168 * 1024;
  vq_encoder_kernel<<<nb, 256, lds_bytes, stream>>>(
      points, N, eW0, eb0, eg0, ebt0, eW1, eb1, eg1, ebt1, eW2, eb2, eg2, ebt2,
      pooled);

  vq_latent_kernel<<<1, 256, 0, stream>>>(pooled, oW, ob, z_e_out);
  vq_argmin_kernel<<<8192 / 256, 256, 0, stream>>>(z_e_out, codebook, key);
  vq_decoder_mlp_kernel<<<1, 512, 0, stream>>>(key, codebook, z_q_out,
      dW0, db0, dg0, dbt0, dW1, db1, dg1, dbt1, dW2, db2, dg2, dbt2, g_ws);
  vq_final_kernel<<<KO / 256, 256, 0, stream>>>(g_ws, doW, dob, recon, KO);
}
